// MultiHeadSelfAttention_15573551415864
// MI455X (gfx1250) — matrix operand order repaired and checked by reading, not compiled, not run on hardware
//
#include <hip/hip_runtime.h>
#include <hip/hip_bf16.h>

// ---------------------------------------------------------------------------
// MHA + RoPE for MI455X (gfx1250), bf16 WMMA compute / fp32 accumulate.
// ---------------------------------------------------------------------------

typedef __attribute__((ext_vector_type(16))) __bf16 v16bf;
typedef __attribute__((ext_vector_type(8)))  float  v8f;

union Frag {
    v16bf v;
    uint4 q[2];
    unsigned short us[16];
};

#define BATCH 2
#define SEQ   2048
#define DM    1024
#define NH    16
#define DK    64
#define BS    (BATCH * SEQ)   // 4096
#define E3    (3 * DM)        // 3072

__device__ __forceinline__ unsigned short f2bf(float f) {
    unsigned int u = __builtin_bit_cast(unsigned int, f);
    u += 0x7fffu + ((u >> 16) & 1u);           // round-to-nearest-even
    return (unsigned short)(u >> 16);
}

// A fragment (16x32 bf16): two 16B runs 16 ushorts apart (p pre-offset by half*8)
__device__ __forceinline__ void loadA(Frag& f, const unsigned short* p) {
    f.q[0] = *(const uint4*)(p);
    f.q[1] = *(const uint4*)(p + 16);
}
// B fragment (32x16 bf16): two 16B runs 16 ushorts apart (p pre-offset by half*8)
__device__ __forceinline__ void loadB(Frag& f, const unsigned short* p) {
    f.q[0] = *(const uint4*)(p);
    f.q[1] = *(const uint4*)(p + 16);
}

__device__ __forceinline__ v8f wmma_bf16(const Frag& a, const Frag& b, v8f c) {
    return __builtin_amdgcn_wmma_f32_16x16x32_bf16(
        false, a.v, false, b.v, (short)0, c, false, false);
}

// ---------------------------------------------------------------------------
// fp32 -> bf16 conversion (vectorized x4)
// ---------------------------------------------------------------------------
__global__ void cvt_f32_bf16(const float4* __restrict__ src,
                             unsigned short* __restrict__ dst, int n4) {
    int i = blockIdx.x * blockDim.x + threadIdx.x;
    if (i < n4) {
        float4 v = src[i];
        union { unsigned short u[4]; uint2 d; } o;
        o.u[0] = f2bf(v.x); o.u[1] = f2bf(v.y);
        o.u[2] = f2bf(v.z); o.u[3] = f2bf(v.w);
        *(uint2*)(dst + 4 * (size_t)i) = o.d;
    }
}

// ---------------------------------------------------------------------------
// C[M,N] fp32 = A[M,K] bf16  @  W[N,K] bf16 ^T   (both row-major)
// One wave computes a 32x64 output tile; k-loop software double-buffered.
// ---------------------------------------------------------------------------
__global__ __launch_bounds__(256) void gemm_bf16_nt(
        const unsigned short* __restrict__ A,
        const unsigned short* __restrict__ W,
        float* __restrict__ C, int M, int N, int K) {
    int lane = threadIdx.x & 31;
    int wave = (blockIdx.x * blockDim.x + threadIdx.x) >> 5;
    int nt64 = N >> 6;
    int mt = wave / nt64;              // 32-row tile index
    int nt = wave % nt64;              // 64-col tile index
    if (mt * 32 >= M) return;

    int n    = lane & 15;
    int half = lane >> 4;

    v8f acc[2][4];
    #pragma unroll
    for (int m2 = 0; m2 < 2; ++m2)
        #pragma unroll
        for (int t = 0; t < 4; ++t)
            #pragma unroll
            for (int r = 0; r < 8; ++r) acc[m2][t][r] = 0.0f;

    const unsigned short* arow0 = A + (size_t)(mt * 32 + n) * K + half * 8;
    const unsigned short* arow1 = arow0 + (size_t)16 * K;
    const unsigned short* wcol[4];
    #pragma unroll
    for (int t = 0; t < 4; ++t)
        wcol[t] = W + (size_t)(nt * 64 + t * 16 + n) * K + half * 8;

    Frag a0[2], a1[2], b0[4], b1[4];
    loadA(a0[0], arow0);
    loadA(a0[1], arow1);
    #pragma unroll
    for (int t = 0; t < 4; ++t) loadB(b0[t], wcol[t]);

    for (int kb = 0; kb < K; kb += 64) {
        // stage kb+32
        loadA(a1[0], arow0 + kb + 32);
        loadA(a1[1], arow1 + kb + 32);
        #pragma unroll
        for (int t = 0; t < 4; ++t) loadB(b1[t], wcol[t] + kb + 32);
        // compute kb
        #pragma unroll
        for (int t = 0; t < 4; ++t) {
            acc[0][t] = wmma_bf16(a0[0], b0[t], acc[0][t]);
            acc[1][t] = wmma_bf16(a0[1], b0[t], acc[1][t]);
        }
        // stage kb+64
        if (kb + 64 < K) {
            loadA(a0[0], arow0 + kb + 64);
            loadA(a0[1], arow1 + kb + 64);
            #pragma unroll
            for (int t = 0; t < 4; ++t) loadB(b0[t], wcol[t] + kb + 64);
        }
        // compute kb+32
        #pragma unroll
        for (int t = 0; t < 4; ++t) {
            acc[0][t] = wmma_bf16(a1[0], b1[t], acc[0][t]);
            acc[1][t] = wmma_bf16(a1[1], b1[t], acc[1][t]);
        }
    }

    #pragma unroll
    for (int m2 = 0; m2 < 2; ++m2)
        #pragma unroll
        for (int t = 0; t < 4; ++t)
            #pragma unroll
            for (int r = 0; r < 8; ++r) {
                int row = mt * 32 + m2 * 16 + r + 8 * half;
                int col = nt * 64 + t * 16 + n;
                C[(size_t)row * N + col] = acc[m2][t][r];
            }
}

// ---------------------------------------------------------------------------
// RoPE on Q,K + split heads + bf16 convert. V stored transposed [B,H,DK,S].
// ---------------------------------------------------------------------------
__global__ void rope_split(const float* __restrict__ qkv,        // [BS][E3]
                           const int* __restrict__ pos,          // [SEQ]
                           unsigned short* __restrict__ Qh,      // [B*H][S][DK]
                           unsigned short* __restrict__ Kh,      // [B*H][S][DK]
                           unsigned short* __restrict__ Vt) {    // [B*H][DK][S]
    int idx = blockIdx.x * blockDim.x + threadIdx.x;     // 2^21 threads
    int i = idx & 31;                // pair index 0..31
    int s = (idx >> 5) & (SEQ - 1);
    int h = (idx >> 16) & (NH - 1);
    int b = idx >> 20;
    if (b >= BATCH) return;

    int row = b * SEQ + s;
    const float* base = qkv + (size_t)row * E3 + h * DK + 2 * i;
    float q1 = base[0],        q2 = base[1];
    float k1 = base[DM],       k2 = base[DM + 1];
    float v1 = base[2 * DM],   v2 = base[2 * DM + 1];

    float p = (float)pos[s];
    float freq = __powf(10000.0f, -(float)(2 * i) / (float)DK);
    float ang = p * freq;
    float cs = __cosf(ang), sn = __sinf(ang);

    size_t ho = (((size_t)(b * NH + h)) * SEQ + s) * DK + 2 * i;
    Qh[ho]     = f2bf(q1 * cs - q2 * sn);
    Qh[ho + 1] = f2bf(q1 * sn + q2 * cs);
    Kh[ho]     = f2bf(k1 * cs - k2 * sn);
    Kh[ho + 1] = f2bf(k1 * sn + k2 * cs);

    size_t vo = (((size_t)(b * NH + h)) * DK + 2 * i) * SEQ + s;
    Vt[vo]       = f2bf(v1);
    Vt[vo + SEQ] = f2bf(v2);
}

// ---------------------------------------------------------------------------
// Flash attention: one wave per (b, h, 16-row q tile). Online softmax.
// K and V fragment loads for a 32-key chunk are issued up front; the P->LDS
// round trip then only waits on DS, not on global latency.
// ---------------------------------------------------------------------------
__global__ __launch_bounds__(32) void flash_attn(
        const unsigned short* __restrict__ Qh,
        const unsigned short* __restrict__ Kh,
        const unsigned short* __restrict__ Vt,
        unsigned short* __restrict__ AO) {        // [BS][DM] bf16
    __shared__ __align__(16) unsigned short LP[16 * 32];

    int lane = threadIdx.x & 31;
    int n    = lane & 15;
    int half = lane >> 4;
    int qt = blockIdx.x & (SEQ / 16 - 1);   // 0..127
    int bh = blockIdx.x >> 7;               // 0..31
    int b = bh >> 4, h = bh & 15;
    int qbase = qt * 16;

    const unsigned short* Qp = Qh + (size_t)bh * SEQ * DK;
    const unsigned short* Kp = Kh + (size_t)bh * SEQ * DK;
    const unsigned short* Vp = Vt + (size_t)bh * DK * SEQ;

    // Q fragments (d = 0..31 and 32..63), loaded once
    Frag aq0, aq1;
    {
        const unsigned short* p = Qp + (size_t)(qbase + n) * DK + half * 8;
        loadA(aq0, p);
        loadA(aq1, p + 32);
    }

    v8f accO[4];
    float mrow[8], lrow[8];
    #pragma unroll
    for (int t = 0; t < 4; ++t)
        #pragma unroll
        for (int r = 0; r < 8; ++r) accO[t][r] = 0.0f;
    #pragma unroll
    for (int r = 0; r < 8; ++r) { mrow[r] = -INFINITY; lrow[r] = 0.0f; }

    const float scale_qk = 0.125f;   // 1/sqrt(64)

    int nch = (qbase + 47) >> 5;     // ceil((qbase+16)/32) chunks of 32 keys
    for (int c = 0; c < nch; ++c) {
        int kb = c * 32;
        bool v1ok = (kb + 16) <= (qbase + 15);   // tile 0 is always valid

        // ---- issue all global loads for this chunk up front ----
        Frag bv[4];
        #pragma unroll
        for (int dt = 0; dt < 4; ++dt)
            loadB(bv[dt], Vp + (size_t)(dt * 16 + n) * SEQ + kb + half * 8);

        Frag bk00, bk01, bk10, bk11;
        {
            const unsigned short* kp0 = Kp + (size_t)(kb + n) * DK + half * 8;
            loadB(bk00, kp0);
            loadB(bk01, kp0 + 32);
        }
        if (v1ok) {
            const unsigned short* kp1 = Kp + (size_t)(kb + 16 + n) * DK + half * 8;
            loadB(bk10, kp1);
            loadB(bk11, kp1 + 32);
        }

        // ---- score tiles ----
        float s0[8], s1[8];
        {
            v8f sc;
            #pragma unroll
            for (int r = 0; r < 8; ++r) sc[r] = 0.0f;
            sc = wmma_bf16(aq0, bk00, sc);
            sc = wmma_bf16(aq1, bk01, sc);
            #pragma unroll
            for (int r = 0; r < 8; ++r) {
                int rowq = qbase + r + 8 * half;
                s0[r] = ((kb + n) <= rowq) ? sc[r] * scale_qk : -INFINITY;
            }
        }
        if (v1ok) {
            v8f sc;
            #pragma unroll
            for (int r = 0; r < 8; ++r) sc[r] = 0.0f;
            sc = wmma_bf16(aq0, bk10, sc);
            sc = wmma_bf16(aq1, bk11, sc);
            #pragma unroll
            for (int r = 0; r < 8; ++r) {
                int rowq = qbase + r + 8 * half;
                s1[r] = ((kb + 16 + n) <= rowq) ? sc[r] * scale_qk : -INFINITY;
            }
        } else {
            #pragma unroll
            for (int r = 0; r < 8; ++r) s1[r] = -INFINITY;
        }

        // ---- online softmax update (reduce over the 16 N lanes) ----
        float p0[8], p1[8], scl[8];
        #pragma unroll
        for (int r = 0; r < 8; ++r) {
            float mx = fmaxf(s0[r], s1[r]);
            mx = fmaxf(mx, __shfl_xor(mx, 1, 32));
            mx = fmaxf(mx, __shfl_xor(mx, 2, 32));
            mx = fmaxf(mx, __shfl_xor(mx, 4, 32));
            mx = fmaxf(mx, __shfl_xor(mx, 8, 32));
            float mnew = fmaxf(mrow[r], mx);
            scl[r] = __expf(mrow[r] - mnew);
            p0[r] = __expf(s0[r] - mnew);
            p1[r] = __expf(s1[r] - mnew);
            float rs = p0[r] + p1[r];
            rs += __shfl_xor(rs, 1, 32);
            rs += __shfl_xor(rs, 2, 32);
            rs += __shfl_xor(rs, 4, 32);
            rs += __shfl_xor(rs, 8, 32);
            lrow[r] = lrow[r] * scl[r] + rs;
            mrow[r] = mnew;
        }
        #pragma unroll
        for (int t = 0; t < 4; ++t)
            #pragma unroll
            for (int r = 0; r < 8; ++r) accO[t][r] *= scl[r];

        // ---- P (C layout) -> LDS row-major [m][k] as bf16 ----
        __syncthreads();
        #pragma unroll
        for (int r = 0; r < 8; ++r) {
            int m = r + 8 * half;
            LP[m * 32 + n]      = f2bf(p0[r]);
            LP[m * 32 + 16 + n] = f2bf(p1[r]);
        }
        __syncthreads();

        // ---- A fragment of P from LDS, then P @ V ----
        Frag ap;
        ap.q[0] = *(const uint4*)(LP + n * 32 + half * 8);
        ap.q[1] = *(const uint4*)(LP + n * 32 + half * 8 + 16);

        #pragma unroll
        for (int dt = 0; dt < 4; ++dt)
            accO[dt] = wmma_bf16(ap, bv[dt], accO[dt]);
    }

    // epilogue: divide by row sums, write bf16 [BS][DM]
    #pragma unroll
    for (int dt = 0; dt < 4; ++dt)
        #pragma unroll
        for (int r = 0; r < 8; ++r) {
            float o = accO[dt][r] / lrow[r];
            size_t row = (size_t)(b * SEQ + qbase + r + 8 * half);
            AO[row * DM + h * DK + dt * 16 + n] = f2bf(o);
        }
}

// ---------------------------------------------------------------------------
// Host-side orchestration
// ---------------------------------------------------------------------------
extern "C" void kernel_launch(void* const* d_in, const int* in_sizes, int n_in,
                              void* d_out, int out_size, void* d_ws, size_t ws_size,
                              hipStream_t stream) {
    (void)in_sizes; (void)n_in; (void)out_size; (void)ws_size;

    const float* x    = (const float*)d_in[0];   // [B,S,DM]
    const int*   tpos = (const int*)d_in[1];     // [S]
    const float* wqkv = (const float*)d_in[2];   // [3*DM, DM]
    const float* wo   = (const float*)d_in[3];   // [DM, DM]

    char* ws = (char*)d_ws;
    size_t off = 0;
    unsigned short* xb    = (unsigned short*)(ws + off); off += (size_t)BS * DM * 2;
    unsigned short* wqkvb = (unsigned short*)(ws + off); off += (size_t)E3 * DM * 2;
    unsigned short* wob   = (unsigned short*)(ws + off); off += (size_t)DM * DM * 2;
    float*          qkvf  = (float*)(ws + off);          off += (size_t)BS * E3 * 4;
    unsigned short* Qh    = (unsigned short*)(ws + off); off += (size_t)BS * DM * 2;
    unsigned short* Kh    = (unsigned short*)(ws + off); off += (size_t)BS * DM * 2;
    unsigned short* Vt    = (unsigned short*)(ws + off); off += (size_t)BS * DM * 2;
    unsigned short* AO    = (unsigned short*)(ws + off); off += (size_t)BS * DM * 2;

    // 1) fp32 -> bf16 conversions (x4 vectorized)
    {
        int n1 = BS * DM / 4;
        int n2 = E3 * DM / 4;
        int n3 = DM * DM / 4;
        cvt_f32_bf16<<<(n1 + 255) / 256, 256, 0, stream>>>((const float4*)x, xb, n1);
        cvt_f32_bf16<<<(n2 + 255) / 256, 256, 0, stream>>>((const float4*)wqkv, wqkvb, n2);
        cvt_f32_bf16<<<(n3 + 255) / 256, 256, 0, stream>>>((const float4*)wo, wob, n3);
    }

    // 2) QKV projection: [BS,E3] = xb @ wqkvb^T
    {
        int waves = (BS / 32) * (E3 / 64);   // 6144
        gemm_bf16_nt<<<waves / 8, 256, 0, stream>>>(xb, wqkvb, qkvf, BS, E3, DM);
    }

    // 3) RoPE + head split + V transpose
    {
        int total = BATCH * NH * SEQ * 32;   // 2^21 threads
        rope_split<<<total / 256, 256, 0, stream>>>(qkvf, tpos, Qh, Kh, Vt);
    }

    // 4) Flash attention
    {
        int blocks = BATCH * NH * (SEQ / 16);  // 4096
        flash_attn<<<blocks, 32, 0, stream>>>(Qh, Kh, Vt, AO);
    }

    // 5) Output projection: d_out[BS,DM] = AO @ wob^T
    {
        int waves = (BS / 32) * (DM / 64);   // 2048
        gemm_bf16_nt<<<waves / 8, 256, 0, stream>>>(AO, wob, (float*)d_out, BS, DM, DM);
    }
}
